// SchNet_45569603011351
// MI455X (gfx1250) — matrix operand order repaired and checked by reading, not compiled, not run on hardware
//
#include <hip/hip_runtime.h>
#include <hip/hip_bf16.h>
#include <math.h>

#define N_NODES 50000
#define N_EDGES 625000
#define N_GRAPH 64
#define H       128
#define FLTC    128
#define NG      50
#define NGP     64      // NG padded to multiple of 32 for WMMA K
#define NLAYERS 6
#define CUTOFF  10.0f
#define LOG2F_  0.69314718055994531f
#define NPAD    (N_NODES + 32)   // row-padded so 32-row tiles never go OOB

typedef __bf16 bf16;
typedef __attribute__((ext_vector_type(16))) __bf16 v16bf;
typedef __attribute__((ext_vector_type(8)))  float  v8f;
typedef __attribute__((ext_vector_type(4)))  unsigned int u32x4;
typedef __attribute__((ext_vector_type(8)))  int          i32x8;
typedef __attribute__((ext_vector_type(4)))  int          i32x4;

// native converts -> v_cvt_pk_bf16_f32 / v_cvt_f32_bf16 on gfx1250
__device__ inline bf16 f2bf(float f) { return (bf16)f; }
__device__ inline float bf2f(bf16 b) { return (float)b; }
__device__ inline unsigned short bf2us(bf16 b) { return __builtin_bit_cast(unsigned short, b); }
__device__ inline bf16 us2bf(unsigned short h) { return __builtin_bit_cast(bf16, h); }

__device__ inline float sspf(float x) {        // shifted softplus
  float sp = (x > 20.0f) ? x : log1pf(__expf(x));
  return sp - LOG2F_;
}

// -------------------- CDNA5 TDM helpers (probe-verified builtins) --------------------

__device__ inline void wait_tensor0() {
#if __has_builtin(__builtin_amdgcn_s_wait_tensorcnt)
  __builtin_amdgcn_s_wait_tensorcnt(0);
#else
  asm volatile("s_wait_tensorcnt 0x0" ::: "memory");
#endif
}

#if __has_builtin(__builtin_amdgcn_tensor_load_to_lds)
__device__ inline unsigned lds_offset_of(const void* p) {
  return (unsigned)(unsigned long long)(__attribute__((address_space(3))) const void*)p;
}
// 1-D contiguous tile load Global->LDS via Tensor Data Mover.
// D# layout per CDNA5 ISA:
//  group0: [1:0]=count=1, [63:32]=lds_addr, [120:64]=global_addr, [127:126]=type=2
//  group1: [17:16]=data_size(3=8B), [79:48]=tensor_dim0, [127:112]=tile_dim0,
//          [207:160]=tensor_dim0_stride
__device__ inline void tdm_load_1d(unsigned lds_off, const void* gptr, unsigned bytes) {
  unsigned long long ga = (unsigned long long)gptr;
  unsigned units = bytes >> 3;                   // 8-byte elements
  u32x4 g0 = {};
  g0[0] = 1u;                                    // count=1
  g0[1] = lds_off;                               // lds_addr
  g0[2] = (unsigned)ga;                          // global_addr[31:0]
  g0[3] = (unsigned)((ga >> 32) & 0x01FFFFFFull) | (2u << 30);  // ga[56:32] | type=2
  i32x8 g1 = {};
  g1[0] = (int)(3u << 16);                       // data_size = 8B, wg_mask = 0
  g1[1] = (int)((units & 0xFFFFu) << 16);        // tensor_dim0[15:0]
  g1[2] = (int)((units >> 16) & 0xFFFFu);        // tensor_dim0[31:16]
  g1[3] = (int)((units & 0xFFFFu) << 16);        // tile_dim0
  g1[5] = (int)units;                            // tensor_dim0_stride[31:0]
  i32x4 gz = {};
#if defined(__clang_major__) && (__clang_major__ >= 23)
  i32x8 g4 = {};
  __builtin_amdgcn_tensor_load_to_lds(g0, g1, gz, gz, g4, 0);
#else
  __builtin_amdgcn_tensor_load_to_lds(g0, g1, gz, gz, 0);
#endif
}
#endif

// ---- WMMA operand builders (16x16x32 bf16; layouts per CDNA5 ISA) ----

__device__ inline v16bf load_A_lds(const bf16* As, int ldA, int k0, int lane) {
  v16bf a;
  int row = lane & 15;
  int hi8 = (lane & 16) ? 8 : 0;
  const bf16* base = As + row * ldA + k0;
#pragma unroll
  for (int v = 0; v < 4; ++v) {
    int k = 2 * v + hi8;
    a[2 * v]     = base[k];
    a[2 * v + 1] = base[k + 1];
    a[2 * v + 8] = base[k + 16];
    a[2 * v + 9] = base[k + 17];
  }
  return a;
}

__device__ inline v16bf load_B_packed(const unsigned short* __restrict__ Wp,
                                      int nkc, int w, int kc, int lane) {
  const v16bf* p = (const v16bf*)(Wp + ((size_t)(w * nkc + kc) * 32 + lane) * 16);
  return *p;
}

#define WMMA_BF16(acc, a, b) \
  acc = __builtin_amdgcn_wmma_f32_16x16x32_bf16(false, a, false, b, (short)0, acc, false, false)

// -------------------- prep kernels --------------------

__global__ void prep_edges_kernel(const float* __restrict__ pos,
                                  const int* __restrict__ eidx,
                                  float* __restrict__ dist, float* __restrict__ Cf) {
  int e = blockIdx.x * blockDim.x + threadIdx.x;
  if (e >= N_EDGES) return;
  int r = eidx[e], c = eidx[N_EDGES + e];
  float dx = pos[r * 3 + 0] - pos[c * 3 + 0];
  float dy = pos[r * 3 + 1] - pos[c * 3 + 1];
  float dz = pos[r * 3 + 2] - pos[c * 3 + 2];
  float d = sqrtf(dx * dx + dy * dy + dz * dz);
  dist[e] = d;
  Cf[e] = 0.5f * (__cosf(d * (3.14159265358979f / CUTOFF)) + 1.0f);
}

__global__ void embed_kernel(const int* __restrict__ z, const float* __restrict__ emb,
                             float* __restrict__ v, unsigned short* __restrict__ vbf) {
  int i = blockIdx.x * blockDim.x + threadIdx.x;
  if (i >= N_NODES * H) return;
  float val = emb[z[i >> 7] * H + (i & 127)];
  v[i] = val;
  vbf[i] = bf2us(f2bf(val));
}

// Repack row-major [layers, Ksrc, 128] f32 weights into WMMA-B bf16 layout.
__global__ void pack_B_kernel(const float* __restrict__ src,
                              unsigned short* __restrict__ dst,
                              int Ksrc, int Kdst, int layers) {
  int nkc = Kdst / 32;
  int total = layers * 8 * nkc * 512;
  int i = blockIdx.x * blockDim.x + threadIdx.x;
  if (i >= total) return;
  int j = i & 15;
  int lane = (i >> 4) & 31;
  int rest = i >> 9;               // ((l*8 + w) * nkc) + kc
  int kc = rest % nkc;
  int rest2 = rest / nkc;
  int w = rest2 & 7;
  int l = rest2 >> 3;
  int k = kc * 32 + 2 * ((j & 7) >> 1) + ((lane & 16) ? 8 : 0) + ((j & 8) ? 16 : 0) + (j & 1);
  int n = 16 * w + (lane & 15);
  float val = (k < Ksrc) ? src[((size_t)l * Ksrc + k) * 128 + n] : 0.0f;
  dst[i] = bf2us(f2bf(val));
}

__global__ void zero_kernel(float* __restrict__ p, int n) {
  int i = blockIdx.x * blockDim.x + threadIdx.x;
  if (i < n) p[i] = 0.0f;
}

// ---- vl(bf16) = v(bf16) @ lin_W[l]; A tile staged via TDM (8KB contiguous) ----

__global__ __launch_bounds__(256) void lin_gemm_kernel(const unsigned short* __restrict__ vbf,
                                                       const unsigned short* __restrict__ Wp,
                                                       unsigned short* __restrict__ vl) {
  __shared__ bf16 As[32 * H];
  int tid = threadIdx.x;
  int m0 = blockIdx.x * 32;
  const unsigned short* src = vbf + (size_t)m0 * H;
#if __has_builtin(__builtin_amdgcn_tensor_load_to_lds)
  if (tid < 32) tdm_load_1d(lds_offset_of(As), src, 32 * H * 2);
  wait_tensor0();
  __syncthreads();
#else
  {
    const uint4* s4 = (const uint4*)src;
    uint4* d4 = (uint4*)As;
    for (int i = tid; i < (32 * H) / 8; i += 256) d4[i] = s4[i];
  }
  __syncthreads();
#endif

  int lane = tid & 31, w = tid >> 5;
  int n = 16 * w + (lane & 15);
  int rbase = (lane & 16) ? 8 : 0;
  v8f acc0 = {}, acc1 = {};
#pragma unroll
  for (int kc = 0; kc < 4; ++kc) {
    v16bf b  = load_B_packed(Wp, 4, w, kc, lane);
    v16bf a0 = load_A_lds(As, H, kc * 32, lane);
    v16bf a1 = load_A_lds(As + 16 * H, H, kc * 32, lane);
    WMMA_BF16(acc0, a0, b);
    WMMA_BF16(acc1, a1, b);
  }
#pragma unroll
  for (int r = 0; r < 8; ++r) {
    vl[(size_t)(m0 + rbase + r) * FLTC + n]      = bf2us(f2bf(acc0[r]));
    vl[(size_t)(m0 + 16 + rbase + r) * FLTC + n] = bf2us(f2bf(acc1[r]));
  }
}

// --- fused edge pipeline (32 edges/block): gaussian emb -> MLP(ssp) -> *C
//     -> *vl[row] -> atomicAdd agg[col]. Invalid edges carry C=0 => no guards. ---

__global__ __launch_bounds__(256) void edge_kernel(const float* __restrict__ dist,
                                                   const float* __restrict__ Cf,
                                                   const int* __restrict__ eidx,
                                                   const unsigned short* __restrict__ W1p,
                                                   const float* __restrict__ b1,
                                                   const unsigned short* __restrict__ W2p,
                                                   const float* __restrict__ b2,
                                                   const unsigned short* __restrict__ vl,
                                                   float* __restrict__ agg) {
  __shared__ bf16 Aemb[32 * NGP];
  __shared__ bf16 Mid[32 * FLTC];
  __shared__ float dS[32], cS[32];
  __shared__ int rowS[32], colS[32];

  int tid = threadIdx.x;
  int e0 = blockIdx.x * 32;
  if (tid < 32) {
    int e = e0 + tid;
    bool ok = e < N_EDGES;
    int ri = ok ? eidx[e] : 0;
    dS[tid] = ok ? dist[e] : 1e9f;
    cS[tid] = ok ? Cf[e] : 0.0f;     // C=0 kills padded edges' contribution
    rowS[tid] = ri;
    colS[tid] = ok ? eidx[N_EDGES + e] : 0;
    __builtin_prefetch(vl + (size_t)ri * FLTC, 0, 3);  // hide gather latency behind GEMMs
  }
  __syncthreads();

  // gaussian embedding: thread -> (edge t, 8 consecutive K values); vectorized stores
  {
    const float step = CUTOFF / (float)(NG - 1);
    const float coeff = -0.5f / (step * step);
    int t = tid >> 3;              // 0..31
    int g0 = (tid & 7) * 8;        // 0,8,...,56
    float d = dS[t];
    ushort4 lo, hi;
    float e0v, e1v;
#define GAU(gi) ((gi) < NG ? __expf(coeff * (d - step * (float)(gi)) * (d - step * (float)(gi))) : 0.0f)
    e0v = GAU(g0 + 0); e1v = GAU(g0 + 1); lo.x = bf2us(f2bf(e0v)); lo.y = bf2us(f2bf(e1v));
    e0v = GAU(g0 + 2); e1v = GAU(g0 + 3); lo.z = bf2us(f2bf(e0v)); lo.w = bf2us(f2bf(e1v));
    e0v = GAU(g0 + 4); e1v = GAU(g0 + 5); hi.x = bf2us(f2bf(e0v)); hi.y = bf2us(f2bf(e1v));
    e0v = GAU(g0 + 6); e1v = GAU(g0 + 7); hi.z = bf2us(f2bf(e0v)); hi.w = bf2us(f2bf(e1v));
#undef GAU
    ushort4* dst = (ushort4*)(Aemb + t * NGP + g0);
    dst[0] = lo;
    dst[1] = hi;
  }
  __syncthreads();

  int lane = tid & 31, w = tid >> 5;
  int n = 16 * w + (lane & 15);
  int rbase = (lane & 16) ? 8 : 0;

  // GEMM1: [32,64] x [64,128]
  v8f acc0 = {}, acc1 = {};
#pragma unroll
  for (int kc = 0; kc < 2; ++kc) {
    v16bf b  = load_B_packed(W1p, 2, w, kc, lane);
    v16bf a0 = load_A_lds(Aemb, NGP, kc * 32, lane);
    v16bf a1 = load_A_lds(Aemb + 16 * NGP, NGP, kc * 32, lane);
    WMMA_BF16(acc0, a0, b);
    WMMA_BF16(acc1, a1, b);
  }
  float bias1 = b1[n];
#pragma unroll
  for (int r = 0; r < 8; ++r) {
    Mid[(rbase + r) * FLTC + n]      = f2bf(sspf(acc0[r] + bias1));
    Mid[(16 + rbase + r) * FLTC + n] = f2bf(sspf(acc1[r] + bias1));
  }
  __syncthreads();

  // GEMM2: [32,128] x [128,128]
  v8f c0 = {}, c1 = {};
#pragma unroll
  for (int kc = 0; kc < 4; ++kc) {
    v16bf b  = load_B_packed(W2p, 4, w, kc, lane);
    v16bf a0 = load_A_lds(Mid, FLTC, kc * 32, lane);
    v16bf a1 = load_A_lds(Mid + 16 * FLTC, FLTC, kc * 32, lane);
    WMMA_BF16(c0, a0, b);
    WMMA_BF16(c1, a1, b);
  }
  float bias2 = b2[n];
#pragma unroll
  for (int r = 0; r < 8; ++r) {
    int m = rbase + r;
    float ev0 = (c0[r] + bias2) * cS[m] * bf2f(us2bf(vl[(size_t)rowS[m] * FLTC + n]));
    atomicAdd(&agg[(size_t)colS[m] * FLTC + n], ev0);
    int m2 = m + 16;
    float ev1 = (c1[r] + bias2) * cS[m2] * bf2f(us2bf(vl[(size_t)rowS[m2] * FLTC + n]));
    atomicAdd(&agg[(size_t)colS[m2] * FLTC + n], ev1);
  }
}

// ---- node update (32-node tile): v += ssp(agg@W1+b1)@W2 + b2; refresh vbf ----

__global__ __launch_bounds__(256) void node_update_kernel(const float* __restrict__ agg,
                                                          const unsigned short* __restrict__ W1p,
                                                          const float* __restrict__ b1,
                                                          const unsigned short* __restrict__ W2p,
                                                          const float* __restrict__ b2,
                                                          float* __restrict__ v,
                                                          unsigned short* __restrict__ vbf) {
  __shared__ bf16 As[32 * H];
  __shared__ bf16 Mid[32 * H];
  int tid = threadIdx.x;
  int m0 = blockIdx.x * 32;
  const float4* a4 = (const float4*)(agg + (size_t)m0 * H);
  unsigned short* AsU = (unsigned short*)As;
  for (int i = tid; i < (32 * H) / 4; i += 256) {
    float4 x = a4[i];
    ushort4 o;
    o.x = bf2us(f2bf(x.x));
    o.y = bf2us(f2bf(x.y));
    o.z = bf2us(f2bf(x.z));
    o.w = bf2us(f2bf(x.w));
    *(ushort4*)(AsU + i * 4) = o;
  }
  __syncthreads();

  int lane = tid & 31, w = tid >> 5;
  int n = 16 * w + (lane & 15);
  int rbase = (lane & 16) ? 8 : 0;

  v8f acc0 = {}, acc1 = {};
#pragma unroll
  for (int kc = 0; kc < 4; ++kc) {
    v16bf b  = load_B_packed(W1p, 4, w, kc, lane);
    v16bf a0 = load_A_lds(As, H, kc * 32, lane);
    v16bf a1 = load_A_lds(As + 16 * H, H, kc * 32, lane);
    WMMA_BF16(acc0, a0, b);
    WMMA_BF16(acc1, a1, b);
  }
  float bias1 = b1[n];
#pragma unroll
  for (int r = 0; r < 8; ++r) {
    Mid[(rbase + r) * H + n]      = f2bf(sspf(acc0[r] + bias1));
    Mid[(16 + rbase + r) * H + n] = f2bf(sspf(acc1[r] + bias1));
  }
  __syncthreads();

  v8f c0 = {}, c1 = {};
#pragma unroll
  for (int kc = 0; kc < 4; ++kc) {
    v16bf b  = load_B_packed(W2p, 4, w, kc, lane);
    v16bf a0 = load_A_lds(Mid, H, kc * 32, lane);
    v16bf a1 = load_A_lds(Mid + 16 * H, H, kc * 32, lane);
    WMMA_BF16(c0, a0, b);
    WMMA_BF16(c1, a1, b);
  }
  float bias2 = b2[n];
#pragma unroll
  for (int r = 0; r < 8; ++r) {
    size_t i0 = (size_t)(m0 + rbase + r) * H + n;
    float nv0 = v[i0] + c0[r] + bias2;
    v[i0] = nv0;
    vbf[i0] = bf2us(f2bf(nv0));
    size_t i1 = (size_t)(m0 + 16 + rbase + r) * H + n;
    float nv1 = v[i1] + c1[r] + bias2;
    v[i1] = nv1;
    vbf[i1] = bf2us(f2bf(nv1));
  }
}

// ---- readout: h = ssp(v@uW1+b1)@uW2 + b2; segment-sum over batch ----

__global__ __launch_bounds__(64) void readout_kernel(const float* __restrict__ v,
                                                     const int* __restrict__ batch,
                                                     const float* __restrict__ uW1,
                                                     const float* __restrict__ ub1,
                                                     const float* __restrict__ uW2,
                                                     const float* __restrict__ ub2,
                                                     float* __restrict__ out) {
  __shared__ float red[64];
  int nidx = blockIdx.x, j = threadIdx.x;
  const float* vr = v + (size_t)nidx * H;
  float acc = ub1[j];
#pragma unroll 4
  for (int k = 0; k < H; ++k) acc += vr[k] * uW1[k * 64 + j];
  red[j] = sspf(acc) * uW2[j];
  __syncthreads();
  for (int s = 32; s > 0; s >>= 1) {
    if (j < s) red[j] += red[j + s];
    __syncthreads();
  }
  if (j == 0) atomicAdd(&out[batch[nidx]], red[0] + ub2[0]);
}

// -------------------- host launcher --------------------

extern "C" void kernel_launch(void* const* d_in, const int* in_sizes, int n_in,
                              void* d_out, int out_size, void* d_ws, size_t ws_size,
                              hipStream_t stream) {
  (void)in_sizes; (void)n_in; (void)out_size; (void)ws_size;
  const int*   z      = (const int*)d_in[0];
  const float* pos    = (const float*)d_in[1];
  const int*   batch  = (const int*)d_in[2];
  const int*   eidx   = (const int*)d_in[3];
  const float* emb    = (const float*)d_in[4];
  const float* lin_W  = (const float*)d_in[5];
  const float* mlp_W1 = (const float*)d_in[6];
  const float* mlp_b1 = (const float*)d_in[7];
  const float* mlp_W2 = (const float*)d_in[8];
  const float* mlp_b2 = (const float*)d_in[9];
  const float* v_W1   = (const float*)d_in[10];
  const float* v_b1   = (const float*)d_in[11];
  const float* v_W2   = (const float*)d_in[12];
  const float* v_b2   = (const float*)d_in[13];
  const float* u_W1   = (const float*)d_in[14];
  const float* u_b1   = (const float*)d_in[15];
  const float* u_W2   = (const float*)d_in[16];
  const float* u_b2   = (const float*)d_in[17];
  float* out = (float*)d_out;

  char* ws = (char*)d_ws;
  size_t off = 0;
  auto alloc = [&](size_t bytes) -> void* {
    off = (off + 255) & ~(size_t)255;
    void* p = ws + off;
    off += bytes;
    return p;
  };
  float*          dist = (float*)alloc(sizeof(float) * N_EDGES);
  float*          Cf   = (float*)alloc(sizeof(float) * N_EDGES);
  float*          v    = (float*)alloc(sizeof(float) * NPAD * H);
  unsigned short* vbf  = (unsigned short*)alloc(2ull * NPAD * H);
  unsigned short* vl   = (unsigned short*)alloc(2ull * NPAD * FLTC);
  float*          agg  = (float*)alloc(sizeof(float) * NPAD * FLTC);
  const int STR128 = 8 * 4 * 512;   // packed per-layer elements, K=128
  const int STR64  = 8 * 2 * 512;   // K=64
  unsigned short* linbf = (unsigned short*)alloc(2ull * NLAYERS * STR128);
  unsigned short* w1bf  = (unsigned short*)alloc(2ull * NLAYERS * STR64);
  unsigned short* w2bf  = (unsigned short*)alloc(2ull * NLAYERS * STR128);
  unsigned short* vw1bf = (unsigned short*)alloc(2ull * NLAYERS * STR128);
  unsigned short* vw2bf = (unsigned short*)alloc(2ull * NLAYERS * STR128);

  prep_edges_kernel<<<(N_EDGES + 255) / 256, 256, 0, stream>>>(pos, eidx, dist, Cf);
  embed_kernel<<<(N_NODES * H + 255) / 256, 256, 0, stream>>>(z, emb, v, vbf);

  int np128 = NLAYERS * STR128;
  int np64  = NLAYERS * STR64;
  pack_B_kernel<<<(np128 + 255) / 256, 256, 0, stream>>>(lin_W,  linbf, 128, 128, NLAYERS);
  pack_B_kernel<<<(np64  + 255) / 256, 256, 0, stream>>>(mlp_W1, w1bf,  NG,  NGP, NLAYERS);
  pack_B_kernel<<<(np128 + 255) / 256, 256, 0, stream>>>(mlp_W2, w2bf,  128, 128, NLAYERS);
  pack_B_kernel<<<(np128 + 255) / 256, 256, 0, stream>>>(v_W1,   vw1bf, 128, 128, NLAYERS);
  pack_B_kernel<<<(np128 + 255) / 256, 256, 0, stream>>>(v_W2,   vw2bf, 128, 128, NLAYERS);
  zero_kernel<<<1, 64, 0, stream>>>(out, N_GRAPH);

  int ntile = (N_NODES + 31) / 32;     // 1563 (last tile runs in padding)
  int etile = (N_EDGES + 31) / 32;     // 19532
  for (int l = 0; l < NLAYERS; ++l) {
    lin_gemm_kernel<<<ntile, 256, 0, stream>>>(vbf, linbf + (size_t)l * STR128, vl);
    zero_kernel<<<(N_NODES * FLTC + 255) / 256, 256, 0, stream>>>(agg, N_NODES * FLTC);
    edge_kernel<<<etile, 256, 0, stream>>>(dist, Cf, eidx,
                                           w1bf + (size_t)l * STR64,  mlp_b1 + l * FLTC,
                                           w2bf + (size_t)l * STR128, mlp_b2 + l * FLTC,
                                           vl, agg);
    node_update_kernel<<<ntile, 256, 0, stream>>>(agg,
                                                  vw1bf + (size_t)l * STR128, v_b1 + l * H,
                                                  vw2bf + (size_t)l * STR128, v_b2 + l * H,
                                                  v, vbf);
  }
  readout_kernel<<<N_NODES, 64, 0, stream>>>(v, batch, u_W1, u_b1, u_W2, u_b2, out);
}
